// RNN_63153199120819
// MI455X (gfx1250) — matrix operand order repaired and checked by reading, not compiled, not run on hardware
//
#include <hip/hip_runtime.h>
#include <hip/hip_bf16.h>
#include <stdint.h>

// Problem constants (from reference)
#define TSTEPS 128
#define BSZ    1024
#define LAGD   6
#define HD     256
#define GD     1024            // 4*H
#define MTOT   (TSTEPS*BSZ)    // 131072 rows for the MLP stages

typedef __attribute__((ext_vector_type(16))) __bf16 bf16x16;
typedef __attribute__((ext_vector_type(8)))  float  f32x8;

union Frag { bf16x16 v; uint32_t u[8]; };

__device__ __forceinline__ uint16_t f2bf(float f) {
    uint32_t x = __float_as_uint(f);
    uint32_t r = x + 0x7FFFu + ((x >> 16) & 1u);   // RNE
    return (uint16_t)(r >> 16);
}
__device__ __forceinline__ float bf2f(uint16_t h) {
    return __uint_as_float(((uint32_t)h) << 16);
}
__device__ __forceinline__ float sigm(float x) {
    return 1.f / (1.f + __expf(-x));
}

// ---- CDNA5 WMMA fragment loaders (documented 16-bit layouts, wave32) ----
// A: 16x32 (MxK). lane L holds row M=L%16. half = L/16.
// VGPR v holds bf16 pair at k = (v/4)*16 + half*8 + (v%4)*2.
template<int RELU_IN, typename PTR>
__device__ __forceinline__ Frag load_a(PTR A, int lda, int m0, int k0, int lane) {
    Frag f;
    int m    = m0 + (lane & 15);
    int half = (lane >> 4) & 1;
    PTR base = A + (size_t)m * lda + k0 + (half << 3);
#pragma unroll
    for (int v = 0; v < 8; ++v) {
        int k = ((v >> 2) << 4) + ((v & 3) << 1);
        uint32_t u = *(const uint32_t*)(base + k);
        if (RELU_IN) {                       // zero any bf16 half with sign bit set
            uint32_t neg = (u & 0x80008000u) >> 15;
            u &= ~(neg * 0xFFFFu);
        }
        f.u[v] = u;
    }
    return f;
}
// B: 32x16 (KxN) built from row-major W[N,K]: B[k][n] = W[n0+n][k0+k].
// lane L holds column n = L%16; khalf = L/16; VGPR v holds k = (v/4)*16 + khalf*8 + (v%4)*2, +1.
__device__ __forceinline__ Frag load_b(const uint16_t* __restrict__ W, int ldk,
                                       int n0, int k0, int lane) {
    Frag f;
    int col   = lane & 15;
    int khalf = (lane >> 4) & 1;
    const uint16_t* base = W + (size_t)(n0 + col) * ldk + k0 + (khalf << 3);
#pragma unroll
    for (int v = 0; v < 8; ++v) f.u[v] = *(const uint32_t*)(base + ((v >> 2) << 4) + ((v & 3) << 1));
    return f;
}

// ---------------- utility kernels ----------------
__global__ void k_cvt_bf16(const float* __restrict__ s, uint16_t* __restrict__ d, int n) {
    int i = blockIdx.x * blockDim.x + threadIdx.x;
    if (i < n) d[i] = f2bf(s[i]);
}
__global__ void k_zero_u32(uint32_t* __restrict__ p, int n) {
    int i = blockIdx.x * blockDim.x + threadIdx.x;
    if (i < n) p[i] = 0u;
}

// Input MLP layer 1: out[M,H] = relu(x[M,6] @ Wi1[H,6]^T + b), bf16 out. K=6 -> scalar.
__global__ void k_mlp_in(const float* __restrict__ x, const float* __restrict__ W,
                         const float* __restrict__ b, uint16_t* __restrict__ out) {
    int idx = blockIdx.x * blockDim.x + threadIdx.x;   // over M*H
    int m = idx >> 8, h = idx & 255;
    const float* xr = x + (size_t)m * LAGD;
    const float* wr = W + (size_t)h * LAGD;
    float s = b[h];
#pragma unroll
    for (int i = 0; i < LAGD; ++i) s += xr[i] * wr[i];
    out[idx] = f2bf(s > 0.f ? s : 0.f);
}

// WMMA GEMM, N=K=256 fixed: out[M,256] = act(A[M,256] @ W[256,256]^T + bias)
// block = 256 threads = 8 waves; block covers 64 rows x 256 cols.
template<int RELU_IN, int RELU_OUT>
__global__ __launch_bounds__(256) void k_gemm_n256(
    const uint16_t* __restrict__ A, const uint16_t* __restrict__ W,
    const float* __restrict__ bias, uint16_t* __restrict__ out)
{
    int lane  = threadIdx.x & 31;
    int wv    = threadIdx.x >> 5;
    int m0    = blockIdx.x * 64 + (wv >> 1) * 16;
    int nbase = (wv & 1) * 128;

    f32x8 acc[8];
#pragma unroll
    for (int j = 0; j < 8; ++j)
#pragma unroll
        for (int e = 0; e < 8; ++e) acc[j][e] = 0.f;

    for (int k0 = 0; k0 < HD; k0 += 32) {
        Frag a = load_a<RELU_IN>(A, HD, m0, k0, lane);
#pragma unroll
        for (int j = 0; j < 8; ++j) {
            Frag b = load_b(W, HD, nbase + j * 16, k0, lane);
            acc[j] = __builtin_amdgcn_wmma_f32_16x16x32_bf16(
                false, a.v, false, b.v, (short)0, acc[j], false, false);
        }
    }
    // C layout: col = lane%16, row = (lane/16)*8 + v
    int col = lane & 15;
    int rb  = m0 + ((lane >> 4) << 3);
#pragma unroll
    for (int j = 0; j < 8; ++j) {
        int n = nbase + j * 16 + col;
        float bv = bias[n];
#pragma unroll
        for (int v = 0; v < 8; ++v) {
            float r = acc[j][v] + bv;
            if (RELU_OUT && r < 0.f) r = 0.f;
            out[(size_t)(rb + v) * HD + n] = f2bf(r);
        }
    }
}

// -------- Fused full-timestep LSTM kernel: BOTH layers in one launch. --------
// grid = B/32 blocks; block = 512 threads = 16 waves; each block owns 32 batch rows
// (halves redundant L2 weight traffic vs 16-row blocks; same total wave count).
// Wave wv: row-group rg = wv>>3 (16 rows), col slice = (wv&7)*128.
// Phase 1: gates = u[t]@Wih0^T + h0prev@Whh0^T  -> cell -> h0 (kept in LDS + global)
// Phase 2: gates = h0(LDS)@Wih1^T + h1prev@Whh1^T -> cell -> h1 (hs[t])
__global__ __launch_bounds__(512) void k_lstm2_step(
    const uint16_t* __restrict__ xin,     // u[t]        [B,H] bf16
    const uint16_t* __restrict__ h0prev,  // layer0 h    [B,H] bf16
    float* __restrict__ c0,               // layer0 c    [B,H] f32 (in/out)
    const uint16_t* __restrict__ Wih0, const uint16_t* __restrict__ Whh0,
    const float* __restrict__ bih0, const float* __restrict__ bhh0,
    uint16_t* __restrict__ h0out,         // layer0 h    [B,H] bf16 (out, may alias h0prev)
    const uint16_t* __restrict__ h1prev,  // layer1 h    [B,H] bf16
    float* __restrict__ c1,               // layer1 c    [B,H] f32 (in/out)
    const uint16_t* __restrict__ Wih1, const uint16_t* __restrict__ Whh1,
    const float* __restrict__ bih1, const float* __restrict__ bhh1,
    uint16_t* __restrict__ h1out)         // hs[t]       [B,H] bf16
{
    __shared__ float    gates[32 * GD];   // 128 KB, reused by both layers
    __shared__ uint16_t h0s[32 * HD];     // 16 KB: layer0 h tile feeding layer1

    int lane  = threadIdx.x & 31;
    int wv    = threadIdx.x >> 5;         // 0..15
    int rg    = wv >> 3;                  // 0..1 row group (16 rows each)
    int nbase = (wv & 7) * 128;           // column slice
    int m0    = blockIdx.x * 32 + rg * 16;
    int tid   = threadIdx.x;
    int col   = lane & 15;
    int rbl   = rg * 16 + ((lane >> 4) << 3);   // local C row base (0..31)

    f32x8 acc[8];

    // ================= Layer 0 =================
#pragma unroll
    for (int j = 0; j < 8; ++j)
#pragma unroll
        for (int e = 0; e < 8; ++e) acc[j][e] = 0.f;

#pragma unroll
    for (int k0 = 0; k0 < HD; k0 += 32) {
        Frag a = load_a<0>(xin, HD, m0, k0, lane);
#pragma unroll
        for (int j = 0; j < 8; ++j) {
            Frag b = load_b(Wih0, HD, nbase + j * 16, k0, lane);
            acc[j] = __builtin_amdgcn_wmma_f32_16x16x32_bf16(
                false, a.v, false, b.v, (short)0, acc[j], false, false);
        }
    }
#pragma unroll
    for (int k0 = 0; k0 < HD; k0 += 32) {
        Frag a = load_a<0>(h0prev, HD, m0, k0, lane);
#pragma unroll
        for (int j = 0; j < 8; ++j) {
            Frag b = load_b(Whh0, HD, nbase + j * 16, k0, lane);
            acc[j] = __builtin_amdgcn_wmma_f32_16x16x32_bf16(
                false, a.v, false, b.v, (short)0, acc[j], false, false);
        }
    }
#pragma unroll
    for (int j = 0; j < 8; ++j)
#pragma unroll
        for (int v = 0; v < 8; ++v)
            gates[(rbl + v) * GD + nbase + j * 16 + col] = acc[j][v];
    __syncthreads();

    {   // cell 0: thread owns h-col = tid&255, rows (tid>>8)*16 .. +15
        int hc = tid & 255;
        int r0 = (tid >> 8) * 16;
        float bi_ = bih0[hc]       + bhh0[hc];
        float bf_ = bih0[256 + hc] + bhh0[256 + hc];
        float bg_ = bih0[512 + hc] + bhh0[512 + hc];
        float bo_ = bih0[768 + hc] + bhh0[768 + hc];
#pragma unroll
        for (int rr = 0; rr < 16; ++rr) {
            int r = r0 + rr;
            const float* g = gates + r * GD;
            float gi = g[hc]       + bi_;
            float gf = g[256 + hc] + bf_;
            float gg = g[512 + hc] + bg_;
            float go = g[768 + hc] + bo_;
            size_t off = (size_t)(blockIdx.x * 32 + r) * HD + hc;
            float cn = sigm(gf) * c0[off] + sigm(gi) * tanhf(gg);
            float hn = sigm(go) * tanhf(cn);
            c0[off] = cn;
            uint16_t hb = f2bf(hn);
            h0out[off]       = hb;   // global: next timestep's recurrence
            h0s[r * HD + hc] = hb;   // LDS: feeds layer 1 right now
        }
    }
    __syncthreads();   // h0s ready; gates free for reuse

    // ================= Layer 1 =================
#pragma unroll
    for (int j = 0; j < 8; ++j)
#pragma unroll
        for (int e = 0; e < 8; ++e) acc[j][e] = 0.f;

#pragma unroll
    for (int k0 = 0; k0 < HD; k0 += 32) {
        Frag a = load_a<0>((const uint16_t*)h0s, HD, rg * 16, k0, lane);  // A from LDS
#pragma unroll
        for (int j = 0; j < 8; ++j) {
            Frag b = load_b(Wih1, HD, nbase + j * 16, k0, lane);
            acc[j] = __builtin_amdgcn_wmma_f32_16x16x32_bf16(
                false, a.v, false, b.v, (short)0, acc[j], false, false);
        }
    }
#pragma unroll
    for (int k0 = 0; k0 < HD; k0 += 32) {
        Frag a = load_a<0>(h1prev, HD, m0, k0, lane);
#pragma unroll
        for (int j = 0; j < 8; ++j) {
            Frag b = load_b(Whh1, HD, nbase + j * 16, k0, lane);
            acc[j] = __builtin_amdgcn_wmma_f32_16x16x32_bf16(
                false, a.v, false, b.v, (short)0, acc[j], false, false);
        }
    }
#pragma unroll
    for (int j = 0; j < 8; ++j)
#pragma unroll
        for (int v = 0; v < 8; ++v)
            gates[(rbl + v) * GD + nbase + j * 16 + col] = acc[j][v];
    __syncthreads();

    {   // cell 1
        int hc = tid & 255;
        int r0 = (tid >> 8) * 16;
        float bi_ = bih1[hc]       + bhh1[hc];
        float bf_ = bih1[256 + hc] + bhh1[256 + hc];
        float bg_ = bih1[512 + hc] + bhh1[512 + hc];
        float bo_ = bih1[768 + hc] + bhh1[768 + hc];
#pragma unroll
        for (int rr = 0; rr < 16; ++rr) {
            int r = r0 + rr;
            const float* g = gates + r * GD;
            float gi = g[hc]       + bi_;
            float gf = g[256 + hc] + bf_;
            float gg = g[512 + hc] + bg_;
            float go = g[768 + hc] + bo_;
            size_t off = (size_t)(blockIdx.x * 32 + r) * HD + hc;
            float cn = sigm(gf) * c1[off] + sigm(gi) * tanhf(gg);
            float hn = sigm(go) * tanhf(cn);
            c1[off]    = cn;
            h1out[off] = f2bf(hn);
        }
    }
}

// Output layer 3: out[M,6] = A[M,256] @ Wo3[6,256]^T + bo3 (f32 out)
__global__ void k_mlp_out(const uint16_t* __restrict__ A, const float* __restrict__ W,
                          const float* __restrict__ b, float* __restrict__ out) {
    int m = blockIdx.x * blockDim.x + threadIdx.x;
    if (m >= MTOT) return;
    float s[LAGD];
#pragma unroll
    for (int j = 0; j < LAGD; ++j) s[j] = b[j];
    const uint16_t* ar = A + (size_t)m * HD;
    for (int k = 0; k < HD; ++k) {
        float a = bf2f(ar[k]);
#pragma unroll
        for (int j = 0; j < LAGD; ++j) s[j] += a * W[j * HD + k];
    }
#pragma unroll
    for (int j = 0; j < LAGD; ++j) out[(size_t)m * LAGD + j] = s[j];
}

extern "C" void kernel_launch(void* const* d_in, const int* in_sizes, int n_in,
                              void* d_out, int out_size, void* d_ws, size_t ws_size,
                              hipStream_t stream)
{
    const float* x    = (const float*)d_in[0];
    const float* Wi1  = (const float*)d_in[1];
    const float* bi1  = (const float*)d_in[2];
    const float* Wi2  = (const float*)d_in[3];
    const float* bi2  = (const float*)d_in[4];
    const float* Wi3  = (const float*)d_in[5];
    const float* bi3  = (const float*)d_in[6];
    const float* Wih0 = (const float*)d_in[7];
    const float* Whh0 = (const float*)d_in[8];
    const float* bih0 = (const float*)d_in[9];
    const float* bhh0 = (const float*)d_in[10];
    const float* Wih1 = (const float*)d_in[11];
    const float* Whh1 = (const float*)d_in[12];
    const float* bih1 = (const float*)d_in[13];
    const float* bhh1 = (const float*)d_in[14];
    const float* Wo1  = (const float*)d_in[15];
    const float* bo1  = (const float*)d_in[16];
    const float* Wo2  = (const float*)d_in[17];
    const float* bo2  = (const float*)d_in[18];
    const float* Wo3  = (const float*)d_in[19];
    const float* bo3  = (const float*)d_in[20];
    float* out = (float*)d_out;
    (void)in_sizes; (void)n_in; (void)out_size; (void)ws_size;

    char* ws = (char*)d_ws;
    size_t off = 0;
    auto take = [&](size_t bytes) -> char* {
        char* p = ws + off;
        off += (bytes + 255) & ~(size_t)255;
        return p;
    };
    const size_t bigE = (size_t)MTOT * HD;             // 33.5M elems
    uint16_t* bufA  = (uint16_t*)take(bigE * 2);       // 64 MB
    uint16_t* bufB  = (uint16_t*)take(bigE * 2);       // 64 MB
    uint16_t* hsBuf = (uint16_t*)take(bigE * 2);       // 64 MB
    uint16_t* Wi2b  = (uint16_t*)take((size_t)HD * HD * 2);
    uint16_t* Wi3b  = (uint16_t*)take((size_t)HD * HD * 2);
    uint16_t* Wih0b = (uint16_t*)take((size_t)GD * HD * 2);
    uint16_t* Whh0b = (uint16_t*)take((size_t)GD * HD * 2);
    uint16_t* Wih1b = (uint16_t*)take((size_t)GD * HD * 2);
    uint16_t* Whh1b = (uint16_t*)take((size_t)GD * HD * 2);
    uint16_t* Wo1b  = (uint16_t*)take((size_t)HD * HD * 2);
    uint16_t* Wo2b  = (uint16_t*)take((size_t)HD * HD * 2);
    uint16_t* h0    = (uint16_t*)take((size_t)BSZ * HD * 2);
    uint16_t* zh    = (uint16_t*)take((size_t)BSZ * HD * 2);
    float*    c0    = (float*)   take((size_t)BSZ * HD * 4);
    float*    c1    = (float*)   take((size_t)BSZ * HD * 4);

    // --- init state (zeros): zero-h seed + both cell states ---
    {
        int n = BSZ * HD / 2;  // zh as u32
        k_zero_u32<<<(n + 255) / 256, 256, 0, stream>>>((uint32_t*)zh, n);
        n = BSZ * HD;
        k_zero_u32<<<(n + 255) / 256, 256, 0, stream>>>((uint32_t*)c0, n);
        k_zero_u32<<<(n + 255) / 256, 256, 0, stream>>>((uint32_t*)c1, n);
    }
    // --- weight conversion to bf16 ---
    auto cvt = [&](const float* s, uint16_t* d, int n) {
        k_cvt_bf16<<<(n + 255) / 256, 256, 0, stream>>>(s, d, n);
    };
    cvt(Wi2,  Wi2b,  HD * HD);
    cvt(Wi3,  Wi3b,  HD * HD);
    cvt(Wih0, Wih0b, GD * HD);
    cvt(Whh0, Whh0b, GD * HD);
    cvt(Wih1, Wih1b, GD * HD);
    cvt(Whh1, Whh1b, GD * HD);
    cvt(Wo1,  Wo1b,  HD * HD);
    cvt(Wo2,  Wo2b,  HD * HD);

    // --- input MLP ---
    k_mlp_in<<<(MTOT * HD) / 256, 256, 0, stream>>>(x, Wi1, bi1, bufA);
    k_gemm_n256<0, 1><<<MTOT / 64, 256, 0, stream>>>(bufA, Wi2b, bi2, bufB);
    k_gemm_n256<0, 1><<<MTOT / 64, 256, 0, stream>>>(bufB, Wi3b, bi3, bufA); // bufA = u

    // --- 2-layer LSTM, both layers fused per timestep (128 launches) ---
    for (int t = 0; t < TSTEPS; ++t) {
        const uint16_t* ut  = bufA + (size_t)t * BSZ * HD;
        const uint16_t* h0p = (t == 0) ? zh : h0;
        const uint16_t* h1p = (t == 0) ? zh : (hsBuf + (size_t)(t - 1) * BSZ * HD);
        k_lstm2_step<<<BSZ / 32, 512, 0, stream>>>(
            ut, h0p, c0, Wih0b, Whh0b, bih0, bhh0, h0,
            h1p, c1, Wih1b, Whh1b, bih1, bhh1,
            hsBuf + (size_t)t * BSZ * HD);
    }

    // --- output MLP: relu(hs) folded into A-load of first GEMM ---
    k_gemm_n256<1, 1><<<MTOT / 64, 256, 0, stream>>>(hsBuf, Wo1b, bo1, bufA);
    k_gemm_n256<0, 1><<<MTOT / 64, 256, 0, stream>>>(bufA, Wo2b, bo2, bufB);
    k_mlp_out<<<(MTOT + 255) / 256, 256, 0, stream>>>(bufB, Wo3, bo3, out);
}
